// WEIGHTSPEmbedder3Conv_21062519620291
// MI455X (gfx1250) — matrix operand order repaired and checked by reading, not compiled, not run on hardware
//
#include <hip/hip_runtime.h>
#include <hip/hip_bf16.h>

// ---------------------------------------------------------------------------
// Sizes (fixed by the reference)
// ---------------------------------------------------------------------------
#define BATCH   128
#define NN      1000
#define NE      8000
#define INF     64
#define HIDD    128
#define RDIM    64
#define OUTD    640          // 64 + 3*(64+128)
#define EPS     1e-5f

typedef __attribute__((ext_vector_type(16))) _Float16 v16h;
typedef __attribute__((ext_vector_type(8)))  float    v8f;

__device__ __forceinline__ float lrelu(float v) { return v > 0.0f ? v : 0.01f * v; }

// ---------------------------------------------------------------------------
// Small utility kernels
// ---------------------------------------------------------------------------
__global__ void zero_f32(float* __restrict__ p, long n) {
    long i = (long)blockIdx.x * blockDim.x + threadIdx.x;
    if (i < n) p[i] = 0.0f;
}

// Pack a row-major f32 weight matrix (K x Nc) into per-lane WMMA B-fragment
// order, converted to f16.  Packed index:
//   dst[((kb/32)*tilesN + nt)*512 + lane*16 + e]
//     = W[(kb + (lane>>4)*16 + e) * Nc + nt*16 + (lane&15)]
// so each lane's 16-half fragment is one contiguous 32-byte load.
__global__ void pack_w_f16(const float* __restrict__ src, _Float16* __restrict__ dst,
                           int K, int Nc) {
    int i = blockIdx.x * blockDim.x + threadIdx.x;
    if (i >= K * Nc) return;
    int e    = i & 15;
    int lane = (i >> 4) & 31;
    int blk  = i >> 9;
    int tilesN = Nc >> 4;
    int nt = blk % tilesN;
    int kb = (blk / tilesN) << 5;
    int k  = kb + ((lane >> 4) << 4) + e;
    int c  = (nt << 4) + (lane & 15);
    dst[i] = (_Float16)src[(long)k * Nc + c];
}

__global__ void deg_count(const int* __restrict__ src, const int* __restrict__ dst,
                          float* __restrict__ deg_out, float* __restrict__ deg_in) {
    int e = blockIdx.x * blockDim.x + threadIdx.x;
    if (e < NE) {
        atomicAdd(deg_out + src[e], 1.0f);
        atomicAdd(deg_in  + dst[e], 1.0f);
    }
}

__global__ void deg_norm(const float* __restrict__ deg_out, const float* __restrict__ deg_in,
                         float* __restrict__ nrm_out, float* __restrict__ nrm_in) {
    int i = blockIdx.x * blockDim.x + threadIdx.x;
    if (i < NN) {
        nrm_out[i] = rsqrtf(fmaxf(deg_out[i], 1.0f));
        nrm_in[i]  = rsqrtf(fmaxf(deg_in[i],  1.0f));
    }
}

// ---------------------------------------------------------------------------
// Message-passing scatter: agg[b,dst] += h[b,src] * nrm_out[src] * ew[b,e]*suppl[e]
// (n_in normalization is folded into the GEMM A-row scale)
// One thread per (b, e, group-of-4-features).
// ---------------------------------------------------------------------------
__global__ void scatter_msgs(const float* __restrict__ h, const float* __restrict__ ew,
                             const float* __restrict__ suppl, const int* __restrict__ src,
                             const int* __restrict__ dst, const float* __restrict__ nrm_out,
                             float* __restrict__ agg, int D) {
    long tid = (long)blockIdx.x * blockDim.x + threadIdx.x;
    int  G = D >> 2;
    long total = (long)BATCH * NE * G;
    if (tid >= total) return;
    int g = (int)(tid % G);
    long t = tid / G;
    int e = (int)(t % NE);
    int b = (int)(t / NE);
    int s = src[e], d = dst[e];
    float w = ew[(long)b * NE + e] * suppl[e] * nrm_out[s];
    const float4 hv = *(const float4*)(h + ((long)b * NN + s) * D + g * 4);
    float* ap = agg + ((long)b * NN + d) * D + g * 4;
    atomicAdd(ap + 0, hv.x * w);
    atomicAdd(ap + 1, hv.y * w);
    atomicAdd(ap + 2, hv.z * w);
    atomicAdd(ap + 3, hv.w * w);
}

// ---------------------------------------------------------------------------
// WMMA GEMM, one wave per 16-row M-strip x (NT*16) columns:
//   C[m, coff+n] = act( rowscale(m) * A[m,:K] @ B[K, NT*16] + bias[n] )
// A: f32, converted to f16 inline (loaded ONCE per K-step, reused by NT WMMAs).
// Bp: pre-packed f16 fragments (see pack_w_f16).  NT accumulators stay in VGPRs.
// ---------------------------------------------------------------------------
template<int NT>
__global__ void wmma_gemm(const float* __restrict__ A, const _Float16* __restrict__ Bp,
                          const float* __restrict__ bias, const float* __restrict__ rowscale,
                          float* __restrict__ C, int M, int K, int ldC, long coff, int act) {
    int wave = threadIdx.x >> 5;
    int lane = threadIdx.x & 31;
    int mt = blockIdx.x * (blockDim.x >> 5) + wave;
    if (mt >= (M >> 4)) return;               // whole-wave uniform: EXEC stays all-1s
    int tM = mt << 4;
    int hh  = lane >> 4;                      // 0 or 1 (lane half)
    int l15 = lane & 15;

    int arow = tM + l15;                      // A lane row (lanes 0-15 / 16-31 share M rows)
    float rs = rowscale ? rowscale[arow % NN] : 1.0f;
    const float* abase = A + (long)arow * K;

    v8f acc[NT];
#pragma unroll
    for (int j = 0; j < NT; ++j) acc[j] = {};

    for (int kb = 0; kb < K; kb += 32) {
        // A fragment (16x32 f16): elems 0..15 = K[kb+16h .. +15]
        v16h af;
        const float* ap0 = abase + kb + hh * 16;
        const float* ap1 = ap0 + 8;
#pragma unroll
        for (int e = 0; e < 8; ++e) af[e]     = (_Float16)(ap0[e] * rs);
#pragma unroll
        for (int e = 0; e < 8; ++e) af[8 + e] = (_Float16)(ap1[e] * rs);

        // NT packed B fragments, contiguous 32B per lane; independent WMMA chain
        const _Float16* bbase = Bp + ((long)(kb >> 5) * NT) * 512 + lane * 16;
#pragma unroll
        for (int j = 0; j < NT; ++j) {
            v16h bf = *(const v16h*)(bbase + (long)j * 512);
            acc[j] = __builtin_amdgcn_wmma_f32_16x16x32_f16(false, af, false, bf,
                                                            (short)0, acc[j], false, false);
        }
    }

    // C/D 16x16 f32: lane col = 16j + l15, VGPR r -> row tM + r + 8h
#pragma unroll
    for (int j = 0; j < NT; ++j) {
        int col = j * 16 + l15;
        float bv = bias ? bias[col] : 0.0f;
#pragma unroll
        for (int r = 0; r < 8; ++r) {
            int m = tM + r + 8 * hh;
            float v = acc[j][r] + bv;
            if (act) v = lrelu(v);
            C[(long)m * ldC + coff + col] = v;
        }
    }
}

// ---------------------------------------------------------------------------
// Fused GraphNorm + LeakyReLU (in place) + weighted-mean readout piece.
// One block per batch graph, one thread per feature (HIDD=128).
// ---------------------------------------------------------------------------
__global__ void gnorm_lrelu_wmean(float* __restrict__ h, const float* __restrict__ gamma,
                                  const float* __restrict__ beta, const float* __restrict__ alpha,
                                  const float* __restrict__ gw, const float* __restrict__ ar,
                                  float* __restrict__ out, int outOff) {
    int b = blockIdx.x;
    int f = threadIdx.x;
    const long base = (long)b * NN * HIDD + f;

    float sum = 0.0f;
    for (int n = 0; n < NN; ++n) sum += h[base + (long)n * HIDD];
    float am = alpha[f] * (sum * (1.0f / NN));

    float vs = 0.0f;
    for (int n = 0; n < NN; ++n) { float t = h[base + (long)n * HIDD] - am; vs += t * t; }
    float inv = rsqrtf(vs * (1.0f / NN) + EPS);

    float g = gamma[f], be = beta[f];
    float ws = 0.0f;
    for (int n = 0; n < NN; ++n) {
        float t = h[base + (long)n * HIDD] - am;
        float v = lrelu(g * t * inv + be);
        h[base + (long)n * HIDD] = v;
        ws += gw[(long)b * NN + n] * ar[n] * v;
    }
    out[(long)b * OUTD + outOff + f] = ws * (1.0f / NN);
}

// weighted mean of raw input features (piece 0, width INF=64)
__global__ void wmean_x(const float* __restrict__ x, const float* __restrict__ gw,
                        const float* __restrict__ ar, float* __restrict__ out) {
    int b = blockIdx.x;
    int f = threadIdx.x;   // 0..63
    float ws = 0.0f;
    for (int n = 0; n < NN; ++n)
        ws += gw[(long)b * NN + n] * ar[n] * x[((long)b * NN + n) * INF + f];
    out[(long)b * OUTD + f] = ws * (1.0f / NN);
}

// column sum over nodes: z[b,f] = sum_n P[b,n,f]
__global__ void colsum_nodes(const float* __restrict__ P, float* __restrict__ z) {
    int b = blockIdx.x;
    int f = threadIdx.x;   // 0..127
    float s = 0.0f;
    for (int n = 0; n < NN; ++n) s += P[((long)b * NN + n) * HIDD + f];
    z[(long)b * HIDD + f] = s;
}

// final LeakyReLU over the whole concatenated output
__global__ void final_lrelu(float* __restrict__ out, int n) {
    int i = blockIdx.x * blockDim.x + threadIdx.x;
    if (i < n) out[i] = lrelu(out[i]);
}

// ---------------------------------------------------------------------------
// Host orchestration
// ---------------------------------------------------------------------------
extern "C" void kernel_launch(void* const* d_in, const int* in_sizes, int n_in,
                              void* d_out, int out_size, void* d_ws, size_t ws_size,
                              hipStream_t stream) {
    const float* node_feats = (const float*)d_in[0];
    const float* ew         = (const float*)d_in[1];
    const float* gw         = (const float*)d_in[2];
    const float* W1         = (const float*)d_in[3];
    const float* W2         = (const float*)d_in[4];
    const float* W3         = (const float*)d_in[5];
    const float* suppl      = (const float*)d_in[6];   // (3,E)
    const float* AR         = (const float*)d_in[7];   // (4,N)
    const float* gn_gamma   = (const float*)d_in[8];   // (3,H)
    const float* gn_beta    = (const float*)d_in[9];
    const float* gn_alpha   = (const float*)d_in[10];
    const float* phi_W      = (const float*)d_in[11];  // (3,H,H)
    const float* phi_b      = (const float*)d_in[12];  // (3,H)
    const float* rho_W      = (const float*)d_in[13];  // (3,H,RDIM)
    const float* rho_b      = (const float*)d_in[14];  // (3,RDIM)
    const int*   src        = (const int*)d_in[15];
    const int*   dst        = (const int*)d_in[16];
    float* out = (float*)d_out;

    // --- workspace layout (f16 pack buffers kept 32B-aligned for v16h loads) ---
    const size_t BIG = (size_t)BATCH * NN * HIDD * sizeof(float);   // 65,536,000 B
    float* bufA = (float*)d_ws;                                      // scatter agg / phi out
    float* bufB = (float*)((char*)d_ws + BIG);                       // h (even layers)
    float* bufC = (float*)((char*)d_ws + 2 * BIG);                   // h (odd layers)
    char* sm = (char*)d_ws + 3 * BIG;
    float* z        = (float*)sm; sm += (size_t)BATCH * HIDD * 4;    // 64 KB (keeps alignment)
    _Float16* W1h   = (_Float16*)sm; sm += (size_t)INF  * HIDD * 2;
    _Float16* W2h   = (_Float16*)sm; sm += (size_t)HIDD * HIDD * 2;
    _Float16* W3h   = (_Float16*)sm; sm += (size_t)HIDD * HIDD * 2;
    _Float16* phiWh = (_Float16*)sm; sm += (size_t)3 * HIDD * HIDD * 2;
    _Float16* rhoWh = (_Float16*)sm; sm += (size_t)3 * HIDD * RDIM * 2;
    float* deg_out  = (float*)sm; sm += NN * 4;
    float* deg_in   = (float*)sm; sm += NN * 4;
    float* nrm_out  = (float*)sm; sm += NN * 4;
    float* nrm_in   = (float*)sm; sm += NN * 4;

    // --- degree normalization (shared topology) ---
    zero_f32<<<(2 * NN + 255) / 256, 256, 0, stream>>>(deg_out, 2 * NN); // deg_out+deg_in contiguous
    deg_count<<<(NE + 255) / 256, 256, 0, stream>>>(src, dst, deg_out, deg_in);
    deg_norm<<<(NN + 255) / 256, 256, 0, stream>>>(deg_out, deg_in, nrm_out, nrm_in);

    // --- weight packing to f16 WMMA fragment order (once per call) ---
    pack_w_f16<<<(INF * HIDD + 255) / 256, 256, 0, stream>>>(W1, W1h, INF, HIDD);
    pack_w_f16<<<(HIDD * HIDD + 255) / 256, 256, 0, stream>>>(W2, W2h, HIDD, HIDD);
    pack_w_f16<<<(HIDD * HIDD + 255) / 256, 256, 0, stream>>>(W3, W3h, HIDD, HIDD);
    for (int i = 0; i < 3; ++i) {
        pack_w_f16<<<(HIDD * HIDD + 255) / 256, 256, 0, stream>>>(
            phi_W + (long)i * HIDD * HIDD, phiWh + (long)i * HIDD * HIDD, HIDD, HIDD);
        pack_w_f16<<<(HIDD * RDIM + 255) / 256, 256, 0, stream>>>(
            rho_W + (long)i * HIDD * RDIM, rhoWh + (long)i * HIDD * RDIM, HIDD, RDIM);
    }

    // --- piece 0: weighted mean of raw features ---
    wmean_x<<<BATCH, INF, 0, stream>>>(node_feats, gw, AR + 0 * NN, out);

    const _Float16* Wh[3] = {W1h, W2h, W3h};
    const int wmeanOff[3] = {128, 320, 512};
    const int uroOff[3]   = {64, 256, 448};

    const float* hcur = node_feats;
    int Din = INF;
    const int M = BATCH * NN;                 // 128000 rows
    const int mBlocks = (M >> 4) / 4;         // 8000 M-strips -> 2000 blocks of 4 waves

    for (int i = 0; i < 3; ++i) {
        float* agg  = bufA;
        float* gout = (i & 1) ? bufC : bufB;

        // gconv: scatter messages then GEMM with W (n_in folded as row scale)
        long aggN = (long)M * Din;
        zero_f32<<<(unsigned)((aggN + 255) / 256), 256, 0, stream>>>(agg, aggN);
        long sThreads = (long)BATCH * NE * (Din >> 2);
        scatter_msgs<<<(unsigned)((sThreads + 255) / 256), 256, 0, stream>>>(
            hcur, ew, suppl + (long)i * NE, src, dst, nrm_out, agg, Din);
        wmma_gemm<8><<<mBlocks, 128, 0, stream>>>(
            agg, Wh[i], nullptr, nrm_in, gout, M, Din, HIDD, 0, /*act=*/0);

        // GraphNorm + lrelu (in place) + weighted-mean piece
        gnorm_lrelu_wmean<<<BATCH, HIDD, 0, stream>>>(
            gout, gn_gamma + i * HIDD, gn_beta + i * HIDD, gn_alpha + i * HIDD,
            gw, AR + (long)(i + 1) * NN, out, wmeanOff[i]);

        // UniversalReadout: phi GEMM (+bias+lrelu) -> node sum -> rho GEMM into d_out
        wmma_gemm<8><<<mBlocks, 128, 0, stream>>>(
            gout, phiWh + (long)i * HIDD * HIDD, phi_b + i * HIDD, nullptr,
            agg, M, HIDD, HIDD, 0, /*act=*/1);
        colsum_nodes<<<BATCH, HIDD, 0, stream>>>(agg, z);
        wmma_gemm<4><<<(BATCH >> 4) / 4, 128, 0, stream>>>(
            z, rhoWh + (long)i * HIDD * RDIM, rho_b + i * RDIM, nullptr,
            out, BATCH, HIDD, OUTD, uroOff[i], /*act=*/1);

        hcur = gout;
        Din = HIDD;
    }

    final_lrelu<<<(BATCH * OUTD + 255) / 256, 256, 0, stream>>>(out, BATCH * OUTD);
}
